// MultiHeadAttention_88098369176068
// MI455X (gfx1250) — compile-verified, matrix operand order checked by reading, not run on hardware
//
#include <hip/hip_runtime.h>
#include <hip/hip_bf16.h>
#include <stdint.h>

typedef __bf16 bf16_t;
typedef __attribute__((ext_vector_type(2)))  __bf16 v2bf;
typedef __attribute__((ext_vector_type(8)))  __bf16 v8bf;
typedef __attribute__((ext_vector_type(16))) __bf16 v16bf;
typedef __attribute__((ext_vector_type(8)))  float  v8f;

#define DEV __device__ __forceinline__

// fp32 -> bf16 (v_cvt_pk_bf16_f32 when available, manual RNE fallback)
DEV bf16_t f2bf(float f) {
#if __has_builtin(__builtin_amdgcn_cvt_pk_bf16_f32)
  return __builtin_amdgcn_cvt_pk_bf16_f32(f, f)[0];
#else
  uint32_t u = __float_as_uint(f);
  u = (u + 0x7FFFu + ((u >> 16) & 1u)) >> 16;
  uint16_t s = (uint16_t)u;
  bf16_t b;
  __builtin_memcpy(&b, &s, 2);
  return b;
#endif
}

// pack 4 fp32 -> 4 bf16 in a uint2
DEV uint2 f2bf4(float4 f) {
  union { bf16_t h[4]; uint2 u; } o;
#if __has_builtin(__builtin_amdgcn_cvt_pk_bf16_f32)
  v2bf a = __builtin_amdgcn_cvt_pk_bf16_f32(f.x, f.y);
  v2bf b = __builtin_amdgcn_cvt_pk_bf16_f32(f.z, f.w);
  o.h[0] = a[0]; o.h[1] = a[1]; o.h[2] = b[0]; o.h[3] = b[1];
#else
  o.h[0] = f2bf(f.x); o.h[1] = f2bf(f.y); o.h[2] = f2bf(f.z); o.h[3] = f2bf(f.w);
#endif
  return o.u;
}

// v_perm_b32 pack: low/high 16-bit halves of two dwords -> one dword
// result bytes: lo: {r0.b0, r0.b1, r1.b0, r1.b1}; hi: {r0.b2, r0.b3, r1.b2, r1.b3}
DEV uint32_t pack_lo16(uint32_t r0, uint32_t r1) {
  return __builtin_amdgcn_perm(r1, r0, 0x05040100u);
}
DEV uint32_t pack_hi16(uint32_t r0, uint32_t r1) {
  return __builtin_amdgcn_perm(r1, r0, 0x07060302u);
}

// 16-bit A/B fragment: 8 contiguous bf16 at p (k..k+7) + 8 at p+16 (k+16..k+23)
DEV v16bf load_frag16(const bf16_t* p) {
  v8bf lo = *(const v8bf*)p;
  v8bf hi = *(const v8bf*)(p + 16);
  return __builtin_shufflevector(lo, hi, 0, 1, 2, 3, 4, 5, 6, 7,
                                 8, 9, 10, 11, 12, 13, 14, 15);
}

DEV v8f wmma_bf16(v16bf a, v16bf b, v8f c) {
  return __builtin_amdgcn_wmma_f32_16x16x32_bf16(
      false, a, false, b, (short)0, c, false, false);
}

// ---------------------------------------------------------------------------
// Elementwise fp32 -> bf16 (one pass over x and weights; removes all
// conversion VALU from the GEMM hot loops and halves re-read traffic).
// ---------------------------------------------------------------------------
__global__ void __launch_bounds__(256)
cvt_f32_bf16(const float* __restrict__ in, bf16_t* __restrict__ out, int n4) {
  int i = blockIdx.x * 256 + threadIdx.x;
  if (i < n4) {
    float4 f = ((const float4*)in)[i];
    ((uint2*)out)[i] = f2bf4(f);
  }
}

// ---------------------------------------------------------------------------
// Tiled bf16 GEMM: C[M,N] = A[M,K] * W[K,N], both bf16.
// Block tile 128x128, K-step 64, double-buffered LDS, 256 thr = 8 waves,
// wave tile 32x64 -> 16 WMMAs per K-step between single barriers.
//   QKV_OUT: write bf16 in [B, H, T=2048, D=128] layout, else fp32 row-major.
// ---------------------------------------------------------------------------
template <bool QKV_OUT>
__global__ void __launch_bounds__(256)
gemm_bf16_wmma(const bf16_t* __restrict__ Ag, const bf16_t* __restrict__ Wg,
               void* __restrict__ Cp, int M, int N, int K) {
  constexpr int LD = 72;  // row stride, mult of 8 -> 16B-aligned b128 chunks
  __shared__ __align__(16) bf16_t As[2][128][LD];  // [m][k]
  __shared__ __align__(16) bf16_t Bt[2][128][LD];  // [n][k] (transposed)

  const int tid   = threadIdx.x;
  const int lane  = tid & 31;
  const int wave  = tid >> 5;
  const int waveM = wave & 3;
  const int waveN = wave >> 2;
  const int l16   = lane & 15;
  const int kb    = (lane >> 4) << 3;
  const int rbase = (lane >> 4) << 3;
  const int m0 = blockIdx.y * 128;
  const int n0 = blockIdx.x * 128;

  v8f acc[2][4] = {};

  // ---- staging helper (pure bf16 copies; B transposed via v_perm packs) ----
  auto stage = [&](int p, int kk) {
#pragma unroll
    for (int i = 0; i < 4; ++i) {          // A: 128x64 in 8-elem chunks
      int idx = tid + i * 256;
      int row = idx >> 3;
      int c8  = (idx & 7) << 3;
      const bf16_t* src = &Ag[(size_t)(m0 + row) * K + kk + c8];
      *(uint4*)&As[p][row][c8] = *(const uint4*)src;
      if (i == 0) __builtin_prefetch(src + 128, 0, 1);  // next-next K tile
    }
#pragma unroll
    for (int i = 0; i < 2; ++i) {          // W: 64x128, transpose, pack k-pairs
      int idx = tid + i * 256;
      int krp = idx >> 4;                  // k row-pair 0..31
      int c8  = (idx & 15) << 3;           // n offset
      const bf16_t* src0 = &Wg[(size_t)(kk + 2 * krp) * N + n0 + c8];
      uint4 r0 = *(const uint4*)src0;
      uint4 r1 = *(const uint4*)&Wg[(size_t)(kk + 2 * krp + 1) * N + n0 + c8];
      if (i == 0) __builtin_prefetch(src0 + (size_t)64 * N, 0, 1);
      uint32_t a0[4] = {r0.x, r0.y, r0.z, r0.w};
      uint32_t a1[4] = {r1.x, r1.y, r1.z, r1.w};
#pragma unroll
      for (int d = 0; d < 4; ++d) {
        *(uint32_t*)&Bt[p][c8 + 2 * d][2 * krp]     = pack_lo16(a0[d], a1[d]);
        *(uint32_t*)&Bt[p][c8 + 2 * d + 1][2 * krp] = pack_hi16(a0[d], a1[d]);
      }
    }
  };

  stage(0, 0);
  int p = 0;
  for (int kk = 0; kk < K; kk += 64) {
    __syncthreads();                       // buffer p staged; buffer p^1 free
    if (kk + 64 < K) stage(p ^ 1, kk + 64);
#pragma unroll
    for (int ks = 0; ks < 2; ++ks) {
      v16bf a0 = load_frag16(&As[p][waveM * 32 + l16][ks * 32 + kb]);
      v16bf a1 = load_frag16(&As[p][waveM * 32 + 16 + l16][ks * 32 + kb]);
#pragma unroll
      for (int nt = 0; nt < 4; ++nt) {
        v16bf bf = load_frag16(&Bt[p][waveN * 64 + nt * 16 + l16][ks * 32 + kb]);
        acc[0][nt] = wmma_bf16(a0, bf, acc[0][nt]);
        acc[1][nt] = wmma_bf16(a1, bf, acc[1][nt]);
      }
    }
    p ^= 1;
  }

  if constexpr (QKV_OUT) {
    bf16_t* Cb = (bf16_t*)Cp;
    const int H = N >> 7;  // heads; head_dim 128, T 2048
#pragma unroll
    for (int mt = 0; mt < 2; ++mt)
#pragma unroll
      for (int nt = 0; nt < 4; ++nt)
#pragma unroll
        for (int r = 0; r < 8; ++r) {
          int gm = m0 + waveM * 32 + mt * 16 + rbase + r;
          int gn = n0 + waveN * 64 + nt * 16 + l16;
          int bb = gm >> 11;
          int tt = gm & 2047;
          int hh = gn >> 7;
          int dd = gn & 127;
          Cb[(((size_t)bb * H + hh) * 2048 + tt) * 128 + dd] =
              f2bf(acc[mt][nt][r]);
        }
  } else {
    float* Cf = (float*)Cp;
#pragma unroll
    for (int mt = 0; mt < 2; ++mt)
#pragma unroll
      for (int nt = 0; nt < 4; ++nt)
#pragma unroll
        for (int r = 0; r < 8; ++r) {
          int gm = m0 + waveM * 32 + mt * 16 + rbase + r;
          int gn = n0 + waveN * 64 + nt * 16 + l16;
          Cf[(size_t)gm * N + gn] = acc[mt][nt][r];
        }
  }
}

// ---------------------------------------------------------------------------
// Flash attention, GQA causal. Q/K/V bf16 in [B,H,T,128]; out bf16 [B,T,E].
// Workgroup: (b, h, 128 q rows); 8 waves x 16 q rows; KV blocks of 64.
// ---------------------------------------------------------------------------
__global__ void __launch_bounds__(256)
flash_attn_gqa(const bf16_t* __restrict__ Q, const bf16_t* __restrict__ K,
               const bf16_t* __restrict__ V, bf16_t* __restrict__ Aout) {
  constexpr int T = 2048, D = 128, E = 2048, HQ = 16, HKV = 4;
  constexpr float SCALE  = 0.08838834764831845f;  // 128^-0.5
  constexpr float NEGBIG = -1e30f;

  __shared__ __align__(16) bf16_t Ks[64][136];    // [kv][d]
  __shared__ __align__(16) bf16_t Vt[128][72];    // [d][kv] (transposed)
  __shared__ __align__(16) bf16_t Pb[8][16][72];  // per-wave P row-major

  const int tid   = threadIdx.x;
  const int lane  = tid & 31;
  const int wave  = tid >> 5;
  const int l16   = lane & 15;
  const int kb    = (lane >> 4) << 3;
  const int rbase = (lane >> 4) << 3;

  const int qt = blockIdx.x;
  const int h  = blockIdx.y;
  const int b  = blockIdx.z;
  const int q0  = qt * 128;
  const int qr0 = q0 + wave * 16;

  const bf16_t* Qh = Q + ((size_t)(b * HQ  + h)        * T) * D;
  const bf16_t* Kh = K + ((size_t)(b * HKV + (h >> 2)) * T) * D;
  const bf16_t* Vh = V + ((size_t)(b * HKV + (h >> 2)) * T) * D;

  // Q fragments straight from global (lane=row, contiguous d == A layout)
  v16bf qfrag[4];
#pragma unroll
  for (int ks = 0; ks < 4; ++ks)
    qfrag[ks] = load_frag16(&Qh[(size_t)(qr0 + l16) * D + ks * 32 + kb]);

  v8f o[8] = {};
  float m_i[8], l_i[8];
#pragma unroll
  for (int r = 0; r < 8; ++r) { m_i[r] = NEGBIG; l_i[r] = 0.f; }

  const int nb = q0 / 64 + 2;
  for (int j = 0; j < nb; ++j) {
    const int kv0 = j * 64;
    __syncthreads();
    // stage K [64][128] (row-major b128 copy)
#pragma unroll
    for (int i = 0; i < 4; ++i) {
      int idx = tid + i * 256;
      int row = idx >> 4;
      int c8  = (idx & 15) << 3;
      *(uint4*)&Ks[row][c8] = *(const uint4*)&Kh[(size_t)(kv0 + row) * D + c8];
    }
    // stage V transposed -> Vt[d][kv]: v_perm packs kv pairs into b32 stores
#pragma unroll
    for (int i = 0; i < 2; ++i) {
      int idx = tid + i * 256;
      int kvp = idx >> 4;                  // kv row-pair 0..31
      int c8  = (idx & 15) << 3;           // d offset
      uint4 r0 = *(const uint4*)&Vh[(size_t)(kv0 + 2 * kvp) * D + c8];
      uint4 r1 = *(const uint4*)&Vh[(size_t)(kv0 + 2 * kvp + 1) * D + c8];
      uint32_t a0[4] = {r0.x, r0.y, r0.z, r0.w};
      uint32_t a1[4] = {r1.x, r1.y, r1.z, r1.w};
#pragma unroll
      for (int d = 0; d < 4; ++d) {
        *(uint32_t*)&Vt[c8 + 2 * d][2 * kvp]     = pack_lo16(a0[d], a1[d]);
        *(uint32_t*)&Vt[c8 + 2 * d + 1][2 * kvp] = pack_hi16(a0[d], a1[d]);
      }
    }
    __syncthreads();

    // S = Q K^T : 16x64, K-dim 128 (4 WMMA k-steps)
    v8f s[4] = {};
#pragma unroll
    for (int nt = 0; nt < 4; ++nt)
#pragma unroll
      for (int ks = 0; ks < 4; ++ks) {
        v16bf kf = load_frag16(&Ks[nt * 16 + l16][ks * 32 + kb]);
        s[nt] = wmma_bf16(qfrag[ks], kf, s[nt]);
      }

    const bool needmask = (kv0 + 63) > qr0;  // wave-uniform
    float corr[8];
#pragma unroll
    for (int r = 0; r < 8; ++r) {
      const int qg = qr0 + rbase + r;
      float mx = NEGBIG;
#pragma unroll
      for (int nt = 0; nt < 4; ++nt) {
        float v = s[nt][r] * SCALE;
        if (needmask) {
          int kg = kv0 + nt * 16 + l16;
          if (kg > qg) v = NEGBIG;
        }
        s[nt][r] = v;
        mx = fmaxf(mx, v);
      }
#pragma unroll
      for (int off = 1; off < 16; off <<= 1)
        mx = fmaxf(mx, __shfl_xor(mx, off, 32));
      float mn = fmaxf(m_i[r], mx);
      float c  = __expf(m_i[r] - mn);
      m_i[r] = mn;
      float psum = 0.f;
#pragma unroll
      for (int nt = 0; nt < 4; ++nt) {
        float pv = __expf(s[nt][r] - mn);
        psum += pv;
        Pb[wave][rbase + r][nt * 16 + l16] = f2bf(pv);
      }
#pragma unroll
      for (int off = 1; off < 16; off <<= 1)
        psum += __shfl_xor(psum, off, 32);
      l_i[r] = l_i[r] * c + psum;
      corr[r] = c;
    }
#pragma unroll
    for (int nt2 = 0; nt2 < 8; ++nt2)
#pragma unroll
      for (int r = 0; r < 8; ++r)
        o[nt2][r] *= corr[r];

    // same-wave LDS RAW: P stores -> A-frag loads
    asm volatile("s_wait_dscnt 0" ::: "memory");

    // O += P V : K-dim 64 (2 k-steps), N=128 (8 tiles)
#pragma unroll
    for (int ks2 = 0; ks2 < 2; ++ks2) {
      v16bf pf = load_frag16(&Pb[wave][l16][ks2 * 32 + kb]);
#pragma unroll
      for (int nt2 = 0; nt2 < 8; ++nt2) {
        v16bf vf = load_frag16(&Vt[nt2 * 16 + l16][ks2 * 32 + kb]);
        o[nt2] = wmma_bf16(pf, vf, o[nt2]);
      }
    }
  }

#pragma unroll
  for (int r = 0; r < 8; ++r) {
    float inv = 1.0f / l_i[r];
    int tt = qr0 + rbase + r;
#pragma unroll
    for (int nt2 = 0; nt2 < 8; ++nt2) {
      Aout[((size_t)b * T + tt) * E + h * D + nt2 * 16 + l16] =
          f2bf(o[nt2][r] * inv);
    }
  }
}

// ---------------------------------------------------------------------------
extern "C" void kernel_launch(void* const* d_in, const int* in_sizes, int n_in,
                              void* d_out, int out_size, void* d_ws,
                              size_t ws_size, hipStream_t stream) {
  (void)in_sizes; (void)n_in; (void)out_size; (void)ws_size;
  const float* x  = (const float*)d_in[0];  // [2,2048,2048]
  const float* Wq = (const float*)d_in[1];  // [2048,2048]
  const float* Wk = (const float*)d_in[2];  // [2048,512]
  const float* Wv = (const float*)d_in[3];  // [2048,512]
  const float* Wo = (const float*)d_in[4];  // [2048,2048]
  float* out = (float*)d_out;               // [2,2048,2048]

  // bf16 workspace layout (element offsets)
  bf16_t* xb  = (bf16_t*)d_ws;               // 8388608
  bf16_t* Wqb = xb  + (size_t)8388608;       // 4194304
  bf16_t* Wkb = Wqb + (size_t)4194304;       // 1048576
  bf16_t* Wvb = Wkb + (size_t)1048576;       // 1048576
  bf16_t* Wob = Wvb + (size_t)1048576;       // 4194304
  bf16_t* Qb  = Wob + (size_t)4194304;       // 8388608  [2,16,2048,128]
  bf16_t* Kb  = Qb  + (size_t)8388608;       // 2097152  [2,4,2048,128]
  bf16_t* Vb  = Kb  + (size_t)2097152;       // 2097152
  bf16_t* Ab  = Vb  + (size_t)2097152;       // 8388608  [2,2048,2048]
  // total ~76 MB

  dim3 blk(256);
  // one-pass fp32 -> bf16 conversions
  cvt_f32_bf16<<<8192, blk, 0, stream>>>(x,  xb,  2097152);
  cvt_f32_bf16<<<4096, blk, 0, stream>>>(Wq, Wqb, 1048576);
  cvt_f32_bf16<<<1024, blk, 0, stream>>>(Wk, Wkb,  262144);
  cvt_f32_bf16<<<1024, blk, 0, stream>>>(Wv, Wvb,  262144);
  cvt_f32_bf16<<<4096, blk, 0, stream>>>(Wo, Wob, 1048576);
  // Q/K/V projections
  gemm_bf16_wmma<true><<<dim3(16, 32), blk, 0, stream>>>(
      xb, Wqb, Qb, 4096, 2048, 2048);
  gemm_bf16_wmma<true><<<dim3(4, 32), blk, 0, stream>>>(
      xb, Wkb, Kb, 4096, 512, 2048);
  gemm_bf16_wmma<true><<<dim3(4, 32), blk, 0, stream>>>(
      xb, Wvb, Vb, 4096, 512, 2048);
  // attention
  flash_attn_gqa<<<dim3(16, 16, 2), blk, 0, stream>>>(Qb, Kb, Vb, Ab);
  // output projection -> fp32
  gemm_bf16_wmma<false><<<dim3(16, 32), blk, 0, stream>>>(
      Ab, Wob, out, 4096, 2048, 2048);
}
